// MNISTKeyedRNN_30537217474997
// MI455X (gfx1250) — matrix operand order repaired and checked by reading, not compiled, not run on hardware
//
#include <hip/hip_runtime.h>
#include <hip/hip_bf16.h>

// ---------------------------------------------------------------------------
// MNIST keyed LSTM on gfx1250 (MI455X).
// Recurrent GEMM h@W_hh via v_wmma_f32_16x16x32_bf16 (f32 accumulate),
// tiles staged with GLOBAL_LOAD_ASYNC_TO_LDS_B128 (ASYNCcnt), double-buffered
// LDS, x@W_ih (K=28) in exact f32 VALU, LSTM pointwise fused in-block.
// ---------------------------------------------------------------------------

typedef __attribute__((ext_vector_type(16))) __bf16 v16bf;
typedef __attribute__((ext_vector_type(8)))  float  v8f;

union BF16x16 { unsigned u[8]; v16bf v; };

#define BDIM 4096      // batch
#define HDIM 1024      // hidden
#define GDIM 4096      // 4*H
#define TDIM 28
#define DDIM 28
#define KB   8         // key batch

#define BM 64          // batch tile
#define BH 64          // hidden-column tile
#define KC 32          // K chunk (WMMA K)
#define NKC (HDIM / KC)

__device__ __forceinline__ unsigned short f2bf(float f) {
    unsigned u = __float_as_uint(f);
    u += 0x7fffu + ((u >> 16) & 1u);          // round-to-nearest-even
    return (unsigned short)(u >> 16);
}
__device__ __forceinline__ float sigmoidf_(float x) {
    return 1.0f / (1.0f + __expf(-x));
}
// LDS offset of a __shared__ object = low 32 bits of its generic address.
__device__ __forceinline__ unsigned lds_off(const void* p) {
    return (unsigned)(unsigned long long)p;
}
// Async global -> LDS, 16 bytes per active lane. Tracked by ASYNCcnt.
__device__ __forceinline__ void async_b128(unsigned lds, const void* gaddr) {
    asm volatile("global_load_async_to_lds_b128 %0, %1, off"
                 :: "v"(lds), "v"(gaddr) : "memory");
}
__device__ __forceinline__ void wait_async0() {
    asm volatile("s_wait_asynccnt 0x0" ::: "memory");
}

// ---------------------------------------------------------------------------
// W_hh f32 -> bf16 bits, swizzled into 4KB tiles:
//   dst[((c0*32 + kc)*64 + n)*32 + kq], c0 = g>>6, n = g&63, kc = k>>5, kq = k&31
// so each (64-col block, 32-K chunk) tile is contiguous and already in the
// transposed [n][k] layout the WMMA B fragments want.
__global__ __launch_bounds__(256) void cvt_whh_swizzle_kernel(
    const float* __restrict__ src, unsigned short* __restrict__ dst) {
    int idx = blockIdx.x * 256 + threadIdx.x;       // over HDIM*GDIM
    int k = idx >> 12;                              // / 4096
    int g = idx & 4095;
    int c0 = g >> 6, n = g & 63;
    int kc = k >> 5, kq = k & 31;
    size_t o = ((((size_t)c0 * 32 + kc) << 6) + n) * 32 + kq;
    dst[o] = f2bf(src[idx]);
}

__global__ __launch_bounds__(256) void zero_f32_kernel(float* p, int n) {
    int i = blockIdx.x * 256 + threadIdx.x;
    if (i < n) p[i] = 0.0f;
}

// ---------------------------------------------------------------------------
// Key pass (B=8): plain f32, negligible cost.
__global__ __launch_bounds__(256) void key_gates_kernel(
    const float* __restrict__ key, int t,
    const float* __restrict__ Wih, const float* __restrict__ Whh,
    const float* __restrict__ bias,
    const float* __restrict__ hk, float* __restrict__ gk) {
    int idx = blockIdx.x * 256 + threadIdx.x;
    if (idx >= KB * GDIM) return;
    int kb = idx / GDIM, g = idx % GDIM;
    float s = bias[g];
    const float* xr = key + ((size_t)kb * TDIM + t) * DDIM;
    for (int k = 0; k < DDIM; ++k) s += xr[k] * Wih[(size_t)k * GDIM + g];
    const float* hr = hk + (size_t)kb * HDIM;
    for (int k = 0; k < HDIM; ++k) s += hr[k] * Whh[(size_t)k * GDIM + g];
    gk[idx] = s;
}

__global__ __launch_bounds__(256) void key_update_kernel(
    const float* __restrict__ gk, float* __restrict__ hk, float* __restrict__ ck) {
    int idx = blockIdx.x * 256 + threadIdx.x;
    if (idx >= KB * HDIM) return;
    int kb = idx / HDIM, n = idx % HDIM;
    const float* g = gk + (size_t)kb * GDIM;
    float ig = sigmoidf_(g[n]);
    float fg = sigmoidf_(g[HDIM + n]);
    float gg = tanhf(g[2 * HDIM + n]);
    float og = sigmoidf_(g[3 * HDIM + n]);
    float cn = fg * ck[idx] + ig * gg;
    ck[idx] = cn;
    hk[idx] = og * tanhf(cn);
}

__global__ __launch_bounds__(256) void key_mean_kernel(
    const float* __restrict__ hk, const float* __restrict__ ck,
    float* __restrict__ hm, float* __restrict__ cm) {
    int n = blockIdx.x * 256 + threadIdx.x;
    if (n >= HDIM) return;
    float hs = 0.0f, cs = 0.0f;
    for (int kb = 0; kb < KB; ++kb) {
        hs += hk[kb * HDIM + n];
        cs += ck[kb * HDIM + n];
    }
    hm[n] = hs * (1.0f / KB);
    cm[n] = cs * (1.0f / KB);
}

__global__ __launch_bounds__(256) void broadcast_state_kernel(
    const float* __restrict__ hm, const float* __restrict__ cm,
    unsigned short* __restrict__ hbf, float* __restrict__ c) {
    int idx = blockIdx.x * 256 + threadIdx.x;   // exactly B*H threads launched
    int n = idx & (HDIM - 1);
    hbf[idx] = f2bf(hm[n]);
    c[idx] = cm[n];
}

// ---------------------------------------------------------------------------
// Fused LSTM timestep. Block = 256 threads = 8 waves; wave w handles gate
// q=w>>1, column-half nh=w&1 of a 64(batch) x 64(hidden) tile, all 4 gates.
__global__ __launch_bounds__(256) void lstm_step_fused(
    const float* __restrict__ xbase, int t,
    const float* __restrict__ Wih,              // [28][4096] f32
    const unsigned short* __restrict__ WhhT,    // swizzled bf16 tiles
    const float* __restrict__ bias,             // [4096]
    const unsigned short* __restrict__ hbf_in,  // [B][1024] bf16 bits
    float* __restrict__ c,                      // [B][1024] in/out
    float* __restrict__ h,                      // [B][1024] out (f32)
    unsigned short* __restrict__ hbf_out)       // [B][1024] out (bf16)
{
    __shared__ float          Xs[BM][DDIM];          // 7 KB
    __shared__ unsigned short As[2][BM][KC];         // 2 x 4 KB
    __shared__ unsigned short Bs[2][4][BH][KC];      // 2 x 16 KB
    __shared__ float          Gs[4][BM][BH];         // 64 KB

    const int tid  = threadIdx.x;
    const int m0   = blockIdx.x * BM;   // batch row base
    const int cb   = blockIdx.y;        // hidden 64-col block
    const int n0   = cb * BH;
    const int lane = tid & 31;
    const int wid  = tid >> 5;
    const int q    = wid >> 1;          // gate index 0..3 (i,f,g,o)
    const int nh   = wid & 1;           // 32-column half within BH
    const int half = lane >> 4;
    const int l16  = lane & 15;

    const unsigned xsBase = lds_off(&Xs[0][0]);
    const unsigned asBase = lds_off(&As[0][0][0]);
    const unsigned bsBase = lds_off(&Bs[0][0][0][0]);

    // issue async loads for K-chunk kcn into LDS buffer `buf`
    auto issue_chunk = [&](int kcn, int buf) {
        // A tile: 64 rows x 64B of h bf16 -> 256 lanes x 16B
        {
            int r = tid >> 2, ch = tid & 3;
            unsigned lds = asBase + (unsigned)buf * (BM * KC * 2)
                         + (unsigned)r * (KC * 2) + (unsigned)ch * 16u;
            const char* g = (const char*)hbf_in
                          + ((size_t)(m0 + r) * HDIM + (size_t)kcn * KC) * 2
                          + (size_t)ch * 16;
            async_b128(lds, g);
        }
        // B tiles: per gate one contiguous 4KB pre-swizzled tile
        for (int q2 = 0; q2 < 4; ++q2) {
            unsigned lds = bsBase + (unsigned)buf * (4 * BH * KC * 2)
                         + (unsigned)q2 * (BH * KC * 2) + (unsigned)tid * 16u;
            size_t tile = (((size_t)(q2 * 16 + cb)) * NKC + kcn) * (BH * KC * 2);
            async_b128(lds, (const char*)WhhT + tile + (size_t)tid * 16);
        }
    };

    // ---- prologue: async-stage x_t tile (7KB) and K-chunk 0 ----
    for (int i = tid; i < BM * 7; i += 256) {        // 64 rows x 7 x 16B
        int r = i / 7, ch = i % 7;
        unsigned lds = xsBase + (unsigned)r * (DDIM * 4) + (unsigned)ch * 16u;
        const char* g = (const char*)(xbase + ((size_t)(m0 + r) * TDIM + t) * DDIM)
                      + (size_t)ch * 16;
        async_b128(lds, g);
    }
    issue_chunk(0, 0);
    wait_async0();
    __syncthreads();

    // ---- init accumulators: bias + x_t @ W_ih (exact f32) ----
    v8f acc[4][2];
    for (int mi = 0; mi < 4; ++mi) {
        for (int nj = 0; nj < 2; ++nj) {
            int nl   = (nh * 2 + nj) * 16 + l16;
            int gcol = q * HDIM + n0 + nl;           // column in [0,4H)
            float s[8];
            float bb = bias[gcol];
            for (int r = 0; r < 8; ++r) s[r] = bb;
            for (int k = 0; k < DDIM; ++k) {
                float w = Wih[(size_t)k * GDIM + gcol];
                for (int r = 0; r < 8; ++r)
                    s[r] += Xs[mi * 16 + half * 8 + r][k] * w;
            }
            v8f a;
            for (int r = 0; r < 8; ++r) a[r] = s[r];
            acc[mi][nj] = a;
        }
    }

    // ---- K loop: h @ W_hh via WMMA bf16, double-buffered async staging ----
    for (int kc = 0; kc < NKC; ++kc) {
        const int cur = kc & 1;
        if (kc + 1 < NKC) issue_chunk(kc + 1, cur ^ 1);
        if (kc + 2 < NKC) {   // prefetch K+2 weight tile into L2
            size_t tile = (((size_t)((tid & 3) * 16 + cb)) * NKC + (kc + 2))
                        * (BH * KC * 2);
            __builtin_prefetch((const char*)WhhT + tile + (size_t)(tid >> 2) * 64,
                               0, 0);
        }

        // B fragments (32x16): lane l16 = N col; half picks K 0-15/16-31
        BF16x16 bfrag[2];
        for (int nj = 0; nj < 2; ++nj) {
            int n = (nh * 2 + nj) * 16 + l16;
            const unsigned* p = (const unsigned*)&Bs[cur][q][n][half * 16];
            for (int j = 0; j < 8; ++j) bfrag[nj].u[j] = p[j];
        }
        // A fragments (16x32) and 8 WMMAs
        for (int mi = 0; mi < 4; ++mi) {
            BF16x16 afrag;
            int mrow = mi * 16 + l16;
            for (int j = 0; j < 8; ++j) {
                int kb = 2 * j + half * 16;
                afrag.u[j] = *(const unsigned*)&As[cur][mrow][kb];
            }
            for (int nj = 0; nj < 2; ++nj) {
                acc[mi][nj] = __builtin_amdgcn_wmma_f32_16x16x32_bf16(
                    false, afrag.v, false, bfrag[nj].v,
                    (short)0, acc[mi][nj], false, false);
            }
        }

        wait_async0();       // my next-chunk asyncs landed
        __syncthreads();     // everyone's landed; everyone done reading cur
    }

    // ---- scatter gate results to LDS ----
    for (int mi = 0; mi < 4; ++mi)
        for (int nj = 0; nj < 2; ++nj) {
            int nl = (nh * 2 + nj) * 16 + l16;
            for (int r = 0; r < 8; ++r)
                Gs[q][mi * 16 + half * 8 + r][nl] = acc[mi][nj][r];
        }
    __syncthreads();

    // ---- fused LSTM pointwise update (16 elems/thread) ----
    for (int i = tid; i < BM * BH; i += 256) {
        int ml = i >> 6, nl = i & 63;
        size_t gidx = (size_t)(m0 + ml) * HDIM + (n0 + nl);
        float ig = sigmoidf_(Gs[0][ml][nl]);
        float fg = sigmoidf_(Gs[1][ml][nl]);
        float gg = tanhf(Gs[2][ml][nl]);
        float og = sigmoidf_(Gs[3][ml][nl]);
        float cn = fg * c[gidx] + ig * gg;
        float hn = og * tanhf(cn);
        c[gidx]       = cn;
        h[gidx]       = hn;
        hbf_out[gidx] = f2bf(hn);
    }
}

// ---------------------------------------------------------------------------
// Final classifier: out[B][10] = h @ W_cls + b_cls (tiny)
__global__ __launch_bounds__(256) void cls_kernel(
    const float* __restrict__ h, const float* __restrict__ Wcls,
    const float* __restrict__ bcls, float* __restrict__ out) {
    int idx = blockIdx.x * 256 + threadIdx.x;
    if (idx >= BDIM * 10) return;
    int m = idx / 10, n = idx % 10;
    float s = bcls[n];
    const float* hr = h + (size_t)m * HDIM;
    for (int k = 0; k < HDIM; ++k) s += hr[k] * Wcls[k * 10 + n];
    out[idx] = s;
}

// ---------------------------------------------------------------------------
extern "C" void kernel_launch(void* const* d_in, const int* in_sizes, int n_in,
                              void* d_out, int out_size, void* d_ws, size_t ws_size,
                              hipStream_t stream) {
    (void)in_sizes; (void)n_in; (void)out_size; (void)ws_size;
    const float* x     = (const float*)d_in[0];   // [4096,28,28]
    const float* key   = (const float*)d_in[1];   // [8,28,28]
    const float* Wih   = (const float*)d_in[2];   // [28,4096]
    const float* Whh   = (const float*)d_in[3];   // [1024,4096]
    const float* bias  = (const float*)d_in[4];   // [4096]
    const float* Wcls  = (const float*)d_in[5];   // [1024,10]
    const float* bcls  = (const float*)d_in[6];   // [10]
    float* out = (float*)d_out;                   // [4096,10]

    // workspace layout (256B aligned)
    char* ws = (char*)d_ws;
    size_t off = 0;
    auto alloc = [&](size_t bytes) -> char* {
        char* p = ws + off;
        off = (off + bytes + 255) & ~(size_t)255;
        return p;
    };
    unsigned short* WhhT = (unsigned short*)alloc((size_t)HDIM * GDIM * 2); // 8 MB
    unsigned short* hbf0 = (unsigned short*)alloc((size_t)BDIM * HDIM * 2); // 8 MB
    unsigned short* hbf1 = (unsigned short*)alloc((size_t)BDIM * HDIM * 2); // 8 MB
    float* c_st = (float*)alloc((size_t)BDIM * HDIM * 4);                   // 16 MB
    float* h_st = (float*)alloc((size_t)BDIM * HDIM * 4);                   // 16 MB
    float* hk   = (float*)alloc((size_t)KB * HDIM * 4);
    float* ck   = (float*)alloc((size_t)KB * HDIM * 4);
    float* gk   = (float*)alloc((size_t)KB * GDIM * 4);
    float* hm   = (float*)alloc((size_t)HDIM * 4);
    float* cm   = (float*)alloc((size_t)HDIM * 4);

    // 1) W_hh -> swizzled bf16 tiles
    {
        int n = HDIM * GDIM;
        cvt_whh_swizzle_kernel<<<(n + 255) / 256, 256, 0, stream>>>(Whh, WhhT);
    }
    // 2) key pass: zero state, 28 steps, mean, broadcast
    zero_f32_kernel<<<(KB * HDIM + 255) / 256, 256, 0, stream>>>(hk, KB * HDIM);
    zero_f32_kernel<<<(KB * HDIM + 255) / 256, 256, 0, stream>>>(ck, KB * HDIM);
    for (int t = 0; t < TDIM; ++t) {
        key_gates_kernel<<<(KB * GDIM + 255) / 256, 256, 0, stream>>>(
            key, t, Wih, Whh, bias, hk, gk);
        key_update_kernel<<<(KB * HDIM + 255) / 256, 256, 0, stream>>>(gk, hk, ck);
    }
    key_mean_kernel<<<(HDIM + 255) / 256, 256, 0, stream>>>(hk, ck, hm, cm);
    broadcast_state_kernel<<<(BDIM * HDIM) / 256, 256, 0, stream>>>(
        hm, cm, hbf0, c_st);

    // 3) main recurrence: 28 fused WMMA timesteps, h bf16 double-buffered
    dim3 grid(BDIM / BM, HDIM / BH);   // 64 x 16
    unsigned short* hin  = hbf0;
    unsigned short* hout = hbf1;
    for (int t = 0; t < TDIM; ++t) {
        lstm_step_fused<<<grid, 256, 0, stream>>>(
            x, t, Wih, WhhT, bias, hin, c_st, h_st, hout);
        unsigned short* tmp = hin; hin = hout; hout = tmp;
    }

    // 4) classifier
    cls_kernel<<<(BDIM * 10 + 255) / 256, 256, 0, stream>>>(h_st, Wcls, bcls, out);
}
